// MemoryDefense_16518444220442
// MI455X (gfx1250) — matrix operand order repaired and checked by reading, not compiled, not run on hardware
//
#include <hip/hip_runtime.h>

typedef __attribute__((ext_vector_type(16))) __bf16 v16bf;
typedef __attribute__((ext_vector_type(8)))  float  v8f;
typedef __attribute__((address_space(3))) unsigned short lds_us_t;

#define BATCH 512
#define FEAT  5184
#define MEMD  1200
#define MEMP  1216   // padded memory dim (multiple of 64)

// ---- workspace layout (bytes). Decoder temps alias freed encoder buffers. ----
#define WS_A0     ((size_t)0)           // 35.7MB enc1 out (512,16,33,33) | dec tmp2 (512,16,33,33)
#define WS_A1     ((size_t)36000000)    // 18.9MB enc2 out (512,32,17,17) | dec tmp1 (512,32,17,17)
#define WS_A2     ((size_t)55000000)    // 10.6MB z (512,5184)           | z_hat_non_target
#define WS_MEMBF  ((size_t)66000000)    // 12.6MB mem bf16 (1216,5184)   row-major (B of GEMM1)
#define WS_MEMTBF ((size_t)79000000)    // 12.6MB mem^T bf16 (5184,1216) (B of GEMM2)
#define WS_ZBF    ((size_t)92000000)    // 5.3MB z bf16 (512,5184)
#define WS_LOGIT  ((size_t)98000000)    // 2.5MB logits (512,1216) f32
#define WS_W2BF   ((size_t)101000000)   // 1.25MB w2 bf16 (512,1216)
#define WS_MEAN   ((size_t)103000000)   // 64 f32
#define WS_VAR    ((size_t)103001024)   // 64 f32
#define WS_ZN     ((size_t)103004096)   // 512 f32
#define WS_MN     ((size_t)103008192)   // 1200 f32

__device__ __forceinline__ unsigned short f2bf(float f) {
    unsigned int u = __float_as_uint(f);
    unsigned int r = (u + 0x7FFFu + ((u >> 16) & 1u)) >> 16;
    return (unsigned short)r;
}

// ---------------- generic direct conv (stride s, pad p), NCHW, OIHW ----------------
__global__ void conv_fwd(const float* __restrict__ in, const float* __restrict__ w,
                         const float* __restrict__ bias, float* __restrict__ out,
                         int N, int Cin, int Hin, int Win, int Cout, int Hout, int Wout,
                         int k, int stride, int pad) {
    long i = (long)blockIdx.x * blockDim.x + threadIdx.x;
    long total = (long)N * Cout * Hout * Wout;
    if (i >= total) return;
    int x = i % Wout;
    int y = (i / Wout) % Hout;
    int o = (i / ((long)Wout * Hout)) % Cout;
    int n = i / ((long)Wout * Hout * Cout);
    float acc = bias[o];
    for (int ci = 0; ci < Cin; ++ci) {
        const float* inc = in + ((long)(n * Cin + ci) * Hin) * Win;
        const float* wc = w + (((long)o * Cin + ci) * k) * k;
        for (int ky = 0; ky < k; ++ky) {
            int iy = y * stride - pad + ky;
            if (iy < 0 || iy >= Hin) continue;
            for (int kx = 0; kx < k; ++kx) {
                int ix = x * stride - pad + kx;
                if (ix < 0 || ix >= Win) continue;
                acc += inc[(long)iy * Win + ix] * wc[ky * k + kx];
            }
        }
    }
    out[i] = acc;
}

// ---------------- transposed conv (torch ConvTranspose2d), weight (Cin,Cout,k,k) ----------------
__global__ void convt_fwd(const float* __restrict__ in, const float* __restrict__ w,
                          const float* __restrict__ bias, float* __restrict__ out,
                          int N, int Cin, int Hin, int Win, int Cout, int Hout, int Wout,
                          int k, int stride, int pad) {
    long i = (long)blockIdx.x * blockDim.x + threadIdx.x;
    long total = (long)N * Cout * Hout * Wout;
    if (i >= total) return;
    int x = i % Wout;
    int y = (i / Wout) % Hout;
    int o = (i / ((long)Wout * Hout)) % Cout;
    int n = i / ((long)Wout * Hout * Cout);
    float acc = bias[o];
    for (int ky = 0; ky < k; ++ky) {
        int ty = y + pad - ky;
        if (ty % stride != 0) continue;
        int iy = ty / stride;
        if (iy < 0 || iy >= Hin) continue;
        for (int kx = 0; kx < k; ++kx) {
            int tx = x + pad - kx;
            if (tx % stride != 0) continue;
            int ix = tx / stride;
            if (ix < 0 || ix >= Win) continue;
            for (int ci = 0; ci < Cin; ++ci) {
                acc += in[((long)(n * Cin + ci) * Hin + iy) * Win + ix] *
                       w[(((long)ci * Cout + o) * k + ky) * k + kx];
            }
        }
    }
    out[i] = acc;
}

// ---------------- training-mode BN: batch stats (one block per channel) ----------------
__global__ void bn_stats(const float* __restrict__ x, float* __restrict__ mean,
                         float* __restrict__ var, int C, int HW, int N) {
    __shared__ float sh[256], sh2[256];
    int c = blockIdx.x;
    long cnt = (long)N * HW;
    float s = 0.f, s2 = 0.f;
    for (long e = threadIdx.x; e < cnt; e += blockDim.x) {
        long n = e / HW, sp = e % HW;
        float v = x[((long)n * C + c) * HW + sp];
        s += v; s2 += v * v;
    }
    sh[threadIdx.x] = s; sh2[threadIdx.x] = s2; __syncthreads();
    for (int st = 128; st > 0; st >>= 1) {
        if (threadIdx.x < st) { sh[threadIdx.x] += sh[threadIdx.x + st]; sh2[threadIdx.x] += sh2[threadIdx.x + st]; }
        __syncthreads();
    }
    if (threadIdx.x == 0) {
        float m = sh[0] / (float)cnt;
        mean[c] = m;
        var[c] = sh2[0] / (float)cnt - m * m;
    }
}

// act: 0 = lrelu(0.25), 1 = sigmoid, 2 = none (in-place)
__global__ void bn_act(float* __restrict__ x, const float* __restrict__ mean,
                       const float* __restrict__ var, const float* __restrict__ g,
                       const float* __restrict__ b, int C, int HW, long total, int act) {
    long i = (long)blockIdx.x * blockDim.x + threadIdx.x;
    if (i >= total) return;
    int c = (i / HW) % C;
    float v = (x[i] - mean[c]) * rsqrtf(var[c] + 1e-5f) * g[c] + b[c];
    if (act == 0) v = v > 0.f ? v : 0.25f * v;
    else if (act == 1) v = 1.f / (1.f + expf(-v));
    x[i] = v;
}

// ---------------- f32 -> bf16 with zero row padding ----------------
__global__ void to_bf16_pad(const float* __restrict__ src, unsigned short* __restrict__ dst,
                            int R, int Rp, int C) {
    long i = (long)blockIdx.x * blockDim.x + threadIdx.x;
    long total = (long)Rp * C;
    if (i >= total) return;
    long r = i / C, c = i % C;
    dst[i] = (r < R) ? f2bf(src[r * (long)C + c]) : (unsigned short)0;
}

// f32 (R x C) -> bf16 transposed (C x Rp), zero pad cols >= R
__global__ void to_bf16_T(const float* __restrict__ src, unsigned short* __restrict__ dst,
                          int R, int Rp, int C) {
    long i = (long)blockIdx.x * blockDim.x + threadIdx.x;
    long total = (long)C * Rp;
    if (i >= total) return;
    long m = i % Rp, f = i / Rp;
    dst[i] = (m < R) ? f2bf(src[m * (long)C + f]) : (unsigned short)0;
}

// ---------------- row L2-norm (one block per row) ----------------
__global__ void row_norm(const float* __restrict__ x, float* __restrict__ out, int cols) {
    __shared__ float sh[256];
    int r = blockIdx.x;
    float s = 0.f;
    for (int j = threadIdx.x; j < cols; j += blockDim.x) {
        float v = x[(long)r * cols + j];
        s += v * v;
    }
    sh[threadIdx.x] = s; __syncthreads();
    for (int st = 128; st > 0; st >>= 1) {
        if (threadIdx.x < st) sh[threadIdx.x] += sh[threadIdx.x + st];
        __syncthreads();
    }
    if (threadIdx.x == 0) out[r] = sqrtf(sh[0]);
}

// ================= WMMA bf16 GEMM with async B-panel staging through LDS =================
// C[M,N] = A[M,K] * Bn[N,K]^T.  Bn row-major, rows indexed by N, K contiguous.
// Block = 4 waves (one M-tile each) sharing a 64(N) x 32(K) bf16 B-panel per K-step.
// B-panel is double-buffered in LDS and filled with GLOBAL_LOAD_ASYNC_TO_LDS_B128
// (ASYNCcnt-tracked, no VGPR round trip) -> 4x L2 traffic reduction on B + copy/WMMA overlap.

__device__ __forceinline__ unsigned lds_byte_off(const unsigned short* p) {
    return (unsigned)(unsigned long long)(lds_us_t*)(unsigned short*)p;
}
__device__ __forceinline__ void async_g2l_b128(unsigned lds_byte, const unsigned short* g) {
    asm volatile("global_load_async_to_lds_b128 %0, %1, off"
                 :: "v"(lds_byte), "v"(g) : "memory");
}

__device__ __forceinline__ v16bf load_a16(const unsigned short* p) {
    union { v16bf v; float4 f[2]; } u;
    u.f[0] = *reinterpret_cast<const float4*>(p);        // K = h*8 .. h*8+7
    u.f[1] = *reinterpret_cast<const float4*>(p + 16);   // K = 16+h*8 .. 16+h*8+7
    return u.v;
}
__device__ __forceinline__ v16bf load_b16(const unsigned short* p) {
    union { v16bf v; float4 f[2]; } u;
    u.f[0] = *reinterpret_cast<const float4*>(p);        // K = h*8 .. h*8+7
    u.f[1] = *reinterpret_cast<const float4*>(p + 16);   // K = 16+h*8 .. 16+h*8+7
    return u.v;
}

__global__ __launch_bounds__(128) void gemm_bf16_wmma(const unsigned short* __restrict__ A,
                                                      const unsigned short* __restrict__ Bn,
                                                      float* __restrict__ C,
                                                      int ldA, int ldB, int ldC, int K) {
    __shared__ __align__(16) unsigned short tileB[2][64 * 32];  // [buf][row][k] 4KB each

    const int tid  = threadIdx.x;
    const int lane = tid & 31;
    const int wv   = tid >> 5;
    const int m0 = (blockIdx.x * 4 + wv) * 16;
    const int n0 = blockIdx.y * 64;
    const int mi = lane & 15;          // A row / D(N) index within 16-wide subtile
    const int h  = lane >> 4;          // lane half selects K sub-range

    const unsigned short* arow = A + (long)(m0 + mi) * ldA + h * 8;

    // async staging sources: thread `tid` copies 16B chunks `tid` and `tid+128`
    // of the 64x32 panel (row-major, 64B per row): chunk c -> row c>>2, col (c&3)*8.
    const unsigned short* gp0 = Bn + (long)(n0 + (tid >> 2)) * ldB + (tid & 3) * 8;
    const unsigned short* gp1 = gp0 + 32 * (long)ldB;
    const unsigned lb0 = lds_byte_off(&tileB[0][0]) + (unsigned)tid * 16u;  // buf stride 4096
    const int nk = K / 32;

    // prefetch first panel into buf 0
    async_g2l_b128(lb0, gp0);
    async_g2l_b128(lb0 + 2048u, gp1);

    v8f acc0 = {}; v8f acc1 = {}; v8f acc2 = {}; v8f acc3 = {};
    for (int it = 0; it < nk; ++it) {
        const int cur = it & 1;
        if (it + 1 < nk) {
            // prefetch next panel (buffer last read at it-1; barrier below covered it)
            const unsigned lbn = lb0 + (unsigned)((it + 1) & 1) * 4096u;
            async_g2l_b128(lbn,         gp0 + (long)(it + 1) * 32);
            async_g2l_b128(lbn + 2048u, gp1 + (long)(it + 1) * 32);
            asm volatile("s_wait_asynccnt 0x2" ::: "memory");  // current panel resident
        } else {
            asm volatile("s_wait_asynccnt 0x0" ::: "memory");
        }
        __syncthreads();   // panel visible to all waves

        v16bf a = load_a16(arow + (long)it * 32);
        const unsigned short* tb = &tileB[cur][0];
        v16bf vb0 = load_b16(tb + ( 0 + mi) * 32 + h * 8);
        v16bf vb1 = load_b16(tb + (16 + mi) * 32 + h * 8);
        v16bf vb2 = load_b16(tb + (32 + mi) * 32 + h * 8);
        v16bf vb3 = load_b16(tb + (48 + mi) * 32 + h * 8);
        acc0 = __builtin_amdgcn_wmma_f32_16x16x32_bf16(false, a, false, vb0, (short)0, acc0, false, false);
        acc1 = __builtin_amdgcn_wmma_f32_16x16x32_bf16(false, a, false, vb1, (short)0, acc1, false, false);
        acc2 = __builtin_amdgcn_wmma_f32_16x16x32_bf16(false, a, false, vb2, (short)0, acc2, false, false);
        acc3 = __builtin_amdgcn_wmma_f32_16x16x32_bf16(false, a, false, vb3, (short)0, acc3, false, false);

        __syncthreads();   // all waves done reading `cur` before it is re-filled
    }

    // D layout: lane (mi,h), VGPR r -> element (M = m0 + h*8 + r, N = n0 + t*16 + mi)
    for (int r = 0; r < 8; ++r) {
        long rowoff = (long)(m0 + h * 8 + r) * ldC + n0 + mi;
        C[rowoff +  0] = acc0[r];
        C[rowoff + 16] = acc1[r];
        C[rowoff + 32] = acc2[r];
        C[rowoff + 48] = acc3[r];
    }
}

// ---------------- fused scale / softmax / mask / hardshrink / L1-normalize ----------------
__global__ void softmax_masks(float* __restrict__ logits, const float* __restrict__ zn,
                              const float* __restrict__ mn, const int* __restrict__ labels,
                              float* __restrict__ w1, float* __restrict__ w2,
                              unsigned short* __restrict__ w2bf) {
    __shared__ float sh[256];
    const int i = blockIdx.x, t = threadIdx.x;
    float* row = logits + (long)i * MEMP;
    const float zni = zn[i];
    const int L = labels[i];

    // pass 1: cosine scale + row max
    float mx = -3.4e38f;
    for (int j = t; j < MEMD; j += 256) {
        float s = row[j] / fmaxf(zni * mn[j], 1e-8f);
        row[j] = s;
        mx = fmaxf(mx, s);
    }
    sh[t] = mx; __syncthreads();
    for (int s = 128; s > 0; s >>= 1) { if (t < s) sh[t] = fmaxf(sh[t], sh[t + s]); __syncthreads(); }
    mx = sh[0]; __syncthreads();

    // pass 2: exp + sum
    float sum = 0.f;
    for (int j = t; j < MEMD; j += 256) { float e = expf(row[j] - mx); row[j] = e; sum += e; }
    sh[t] = sum; __syncthreads();
    for (int s = 128; s > 0; s >>= 1) { if (t < s) sh[t] += sh[t + s]; __syncthreads(); }
    const float tot = sh[0]; __syncthreads();

    // pass 3: hardshrunk masked L1 sums
    float s1 = 0.f, s2 = 0.f;
    for (int j = t; j < MEMD; j += 256) {
        float wgt = row[j] / tot;
        float v = (wgt > 1e-12f) ? wgt : 0.f;
        if ((j >> 1) == L) s1 += v; else s2 += v;
    }
    sh[t] = s1; __syncthreads();
    for (int s = 128; s > 0; s >>= 1) { if (t < s) sh[t] += sh[t + s]; __syncthreads(); }
    s1 = sh[0]; __syncthreads();
    sh[t] = s2; __syncthreads();
    for (int s = 128; s > 0; s >>= 1) { if (t < s) sh[t] += sh[t + s]; __syncthreads(); }
    s2 = sh[0]; __syncthreads();

    const float inv1 = 1.f / s1, inv2 = 1.f / s2;
    for (int j = t; j < MEMP; j += 256) {
        if (j < MEMD) {
            float wgt = row[j] / tot;
            float v = (wgt > 1e-12f) ? wgt : 0.f;
            bool tgt = (j >> 1) == L;
            float a = tgt ? v * inv1 : 0.f;
            float b = tgt ? 0.f : v * inv2;
            w1[(long)i * MEMD + j] = a;
            w2[(long)i * MEMD + j] = b;
            w2bf[(long)i * MEMP + j] = f2bf(b);
        } else {
            w2bf[(long)i * MEMP + j] = 0;  // zero-padded K tail for GEMM2
        }
    }
}

// ---------------- z_hat_target: only 2 nonzero weights per row ----------------
__global__ void zhat_target(const float* __restrict__ w1, const int* __restrict__ labels,
                            const float* __restrict__ mem, float* __restrict__ out) {
    long i = (long)blockIdx.x * blockDim.x + threadIdx.x;
    long total = (long)BATCH * FEAT;
    if (i >= total) return;
    long f = i % FEAT, n = i / FEAT;
    int L = labels[n];
    float a = w1[n * (long)MEMD + 2 * L];
    float b = w1[n * (long)MEMD + 2 * L + 1];
    out[i] = a * mem[(long)(2 * L) * FEAT + f] + b * mem[(long)(2 * L + 1) * FEAT + f];
}

// =================================================================================
static void run_decoder(const float* zin, float* rec_out, float* d1buf, float* d2buf,
                        float* mean, float* var,
                        const float* d1w, const float* d1b, const float* g1, const float* bb1,
                        const float* d2w, const float* d2b, const float* g2, const float* bb2,
                        const float* d3w, const float* d3b, hipStream_t stream) {
    long t1 = (long)BATCH * 32 * 17 * 17;
    convt_fwd<<<(t1 + 255) / 256, 256, 0, stream>>>(zin, d1w, d1b, d1buf,
        BATCH, 64, 9, 9, 32, 17, 17, 3, 2, 1);
    bn_stats<<<32, 256, 0, stream>>>(d1buf, mean, var, 32, 17 * 17, BATCH);
    bn_act<<<(t1 + 255) / 256, 256, 0, stream>>>(d1buf, mean, var, g1, bb1, 32, 17 * 17, t1, 0);

    long t2 = (long)BATCH * 16 * 33 * 33;
    convt_fwd<<<(t2 + 255) / 256, 256, 0, stream>>>(d1buf, d2w, d2b, d2buf,
        BATCH, 32, 17, 17, 16, 33, 33, 3, 2, 1);
    bn_stats<<<16, 256, 0, stream>>>(d2buf, mean, var, 16, 33 * 33, BATCH);
    bn_act<<<(t2 + 255) / 256, 256, 0, stream>>>(d2buf, mean, var, g2, bb2, 16, 33 * 33, t2, 0);

    long t3 = (long)BATCH * 64 * 64;
    convt_fwd<<<(t3 + 255) / 256, 256, 0, stream>>>(d2buf, d3w, d3b, rec_out,
        BATCH, 16, 33, 33, 1, 64, 64, 2, 2, 1);
}

extern "C" void kernel_launch(void* const* d_in, const int* in_sizes, int n_in,
                              void* d_out, int out_size, void* d_ws, size_t ws_size,
                              hipStream_t stream) {
    // inputs: setup_inputs() dict order
    const float* x      = (const float*)d_in[0];
    const int*   labels = (const int*)d_in[1];
    const float* e1w = (const float*)d_in[2],  *e1b = (const float*)d_in[3];
    const float* g1  = (const float*)d_in[4],  *b1  = (const float*)d_in[5];
    const float* e2w = (const float*)d_in[6],  *e2b = (const float*)d_in[7];
    const float* g2  = (const float*)d_in[8],  *b2  = (const float*)d_in[9];
    const float* e3w = (const float*)d_in[10], *e3b = (const float*)d_in[11];
    const float* g3  = (const float*)d_in[12], *b3  = (const float*)d_in[13];
    const float* mem = (const float*)d_in[14];
    const float* d1w = (const float*)d_in[15], *d1b = (const float*)d_in[16];
    const float* dg1 = (const float*)d_in[17], *db1 = (const float*)d_in[18];
    const float* d2w = (const float*)d_in[19], *d2b = (const float*)d_in[20];
    const float* dg2 = (const float*)d_in[21], *db2 = (const float*)d_in[22];
    const float* d3w = (const float*)d_in[23], *d3b = (const float*)d_in[24];

    char* ws = (char*)d_ws;
    float* enc1  = (float*)(ws + WS_A0);       // (512,16,33,33) ; later dec tmp2
    float* enc2  = (float*)(ws + WS_A1);       // (512,32,17,17) ; later dec tmp1
    float* z     = (float*)(ws + WS_A2);       // (512,5184) ; later z_hat_non_target
    unsigned short* membf  = (unsigned short*)(ws + WS_MEMBF);
    unsigned short* memtbf = (unsigned short*)(ws + WS_MEMTBF);
    unsigned short* zbf    = (unsigned short*)(ws + WS_ZBF);
    float* logits = (float*)(ws + WS_LOGIT);
    unsigned short* w2bf = (unsigned short*)(ws + WS_W2BF);
    float* mean = (float*)(ws + WS_MEAN);
    float* var  = (float*)(ws + WS_VAR);
    float* zn   = (float*)(ws + WS_ZN);
    float* mn   = (float*)(ws + WS_MN);

    float* out   = (float*)d_out;
    float* o_zt  = out;                                   // (512,5184)
    float* o_rx  = o_zt  + (long)BATCH * FEAT;            // (512,1,64,64)
    float* o_rxh = o_rx  + (long)BATCH * 4096;            // (512,1,64,64)
    float* o_w1  = o_rxh + (long)BATCH * 4096;            // (512,1200)
    float* o_w2  = o_w1  + (long)BATCH * MEMD;            // (512,1200)

    // ---- encoder ----
    long t1 = (long)BATCH * 16 * 33 * 33;
    conv_fwd<<<(t1 + 255) / 256, 256, 0, stream>>>(x, e1w, e1b, enc1,
        BATCH, 1, 64, 64, 16, 33, 33, 1, 2, 1);
    bn_stats<<<16, 256, 0, stream>>>(enc1, mean, var, 16, 33 * 33, BATCH);
    bn_act<<<(t1 + 255) / 256, 256, 0, stream>>>(enc1, mean, var, g1, b1, 16, 33 * 33, t1, 0);

    long t2 = (long)BATCH * 32 * 17 * 17;
    conv_fwd<<<(t2 + 255) / 256, 256, 0, stream>>>(enc1, e2w, e2b, enc2,
        BATCH, 16, 33, 33, 32, 17, 17, 3, 2, 1);
    bn_stats<<<32, 256, 0, stream>>>(enc2, mean, var, 32, 17 * 17, BATCH);
    bn_act<<<(t2 + 255) / 256, 256, 0, stream>>>(enc2, mean, var, g2, b2, 32, 17 * 17, t2, 0);

    long t3 = (long)BATCH * 64 * 9 * 9;   // == 512*5184
    conv_fwd<<<(t3 + 255) / 256, 256, 0, stream>>>(enc2, e3w, e3b, z,
        BATCH, 32, 17, 17, 64, 9, 9, 3, 2, 1);
    bn_stats<<<64, 256, 0, stream>>>(z, mean, var, 64, 9 * 9, BATCH);
    bn_act<<<(t3 + 255) / 256, 256, 0, stream>>>(z, mean, var, g3, b3, 64, 9 * 9, t3, 1);

    // ---- bf16 conversions + norms ----
    long tz = (long)BATCH * FEAT;
    to_bf16_pad<<<(tz + 255) / 256, 256, 0, stream>>>(z, zbf, BATCH, BATCH, FEAT);
    long tm = (long)MEMP * FEAT;
    to_bf16_pad<<<(tm + 255) / 256, 256, 0, stream>>>(mem, membf, MEMD, MEMP, FEAT);
    to_bf16_T<<<(tm + 255) / 256, 256, 0, stream>>>(mem, memtbf, MEMD, MEMP, FEAT);
    row_norm<<<BATCH, 256, 0, stream>>>(z, zn, FEAT);
    row_norm<<<MEMD, 256, 0, stream>>>(mem, mn, FEAT);

    // ---- GEMM1: logits[512,1216] = z[512,5184] * mem^T  (WMMA bf16, async LDS B) ----
    gemm_bf16_wmma<<<dim3(BATCH / 64, MEMP / 64), 128, 0, stream>>>(
        zbf, membf, logits, FEAT, FEAT, MEMP, FEAT);

    // ---- softmax + masks + hardshrink + L1 normalize ----
    softmax_masks<<<BATCH, 256, 0, stream>>>(logits, zn, mn, labels, o_w1, o_w2, w2bf);

    // ---- z_hat_target: 2-row weighted sum (m1 has exactly 2 nonzeros/row) ----
    zhat_target<<<(tz + 255) / 256, 256, 0, stream>>>(o_w1, labels, mem, o_zt);

    // ---- GEMM2: z_hat_non_target[512,5184] = w2[512,1216] * mem[1216,5184] (WMMA bf16) ----
    float* zhat_nt = z;  // reuse (z no longer needed)
    gemm_bf16_wmma<<<dim3(BATCH / 64, FEAT / 64), 128, 0, stream>>>(
        w2bf, memtbf, zhat_nt, MEMP, MEMP, FEAT, MEMP);

    // ---- decoders (temps alias freed encoder buffers) ----
    run_decoder(o_zt, o_rx, enc2, enc1, mean, var,
                d1w, d1b, dg1, db1, d2w, d2b, dg2, db2, d3w, d3b, stream);
    run_decoder(zhat_nt, o_rxh, enc2, enc1, mean, var,
                d1w, d1b, dg1, db1, d2w, d2b, dg2, db2, d3w, d3b, stream);
}
